// FakeConv2d_69973607187188
// MI455X (gfx1250) — compile-verified, matrix operand order checked by reading, not run on hardware
//
#include <hip/hip_runtime.h>
#include <hip/hip_bf16.h>

// Problem dims (from reference)
#define Bn   16
#define Cc   128
#define Hh   56
#define Ww   56
#define HWp  (Hh * Ww)          // 3136
#define Ff   9
#define Kd   (Ff * Cc)          // 1152
#define OUTc 512
#define Mtot (Bn * HWp)         // 50176

typedef __attribute__((ext_vector_type(16))) __bf16 v16bf;
typedef __attribute__((ext_vector_type(8)))  __bf16 v8bf;
typedef __attribute__((ext_vector_type(8)))  float  v8f;
typedef __attribute__((ext_vector_type(4)))  int    v4i;

// CDNA5 async global->LDS copy (GLOBAL_LOAD_ASYNC_TO_LDS_B128, ASYNCcnt-tracked)
#if defined(__has_builtin)
#  if __has_builtin(__builtin_amdgcn_global_load_async_to_lds_b128)
#    define HAS_ASYNC_LDS 1
#  endif
#endif
#ifndef HAS_ASYNC_LDS
#  define HAS_ASYNC_LDS 0
#endif

__device__ __forceinline__ void async_copy_b128(const __bf16* g, __bf16* l) {
#if HAS_ASYNC_LDS
    // Toolchain signature: (v4i* global_src, v4i* lds_dst, imm offset, imm cpol)
    __builtin_amdgcn_global_load_async_to_lds_b128(
        (v4i*)(const void*)g, (v4i*)(void*)l, 0, 0);
#else
    *(uint4*)l = *(const uint4*)g;   // transient regs only; cannot spill across compute
#endif
}

__device__ __forceinline__ void wait_async_all() {
#if HAS_ASYNC_LDS
    asm volatile("s_wait_asynccnt 0x0" ::: "memory");
#endif
}

// ---------------------------------------------------------------------------
// Kernel 1: convert w_conv [OUT, K] f32 -> bf16
// ---------------------------------------------------------------------------
__global__ void wconv_to_bf16_kernel(const float* __restrict__ w,
                                     __bf16* __restrict__ o, int n) {
    int i = blockIdx.x * blockDim.x + threadIdx.x;
    if (i < n) o[i] = (__bf16)w[i];
}

// ---------------------------------------------------------------------------
// Kernel 2: fused offset conv + bilinear sampling.
// One thread per pixel m = b*H*W + h*W + w. Writes sampled[M, K] row-major bf16.
// x (25.7MB) fits in the 192MB L2, so the 4x9x128 gathers per pixel are L2 hits.
// ---------------------------------------------------------------------------
__global__ void __launch_bounds__(256)
deform_sample_kernel(const float* __restrict__ x,
                     const float* __restrict__ w_off,
                     const float* __restrict__ b_off,
                     __bf16* __restrict__ smp) {
    int m  = blockIdx.x * 256 + threadIdx.x;       // < Mtot (exact multiple)
    int b  = m / HWp;
    int hw = m - b * HWp;
    int h  = hw / Ww;
    int w  = hw - h * Ww;
    const float* xb = x + (size_t)b * Cc * HWp;

    // 18 offset dot products over C (bias included)
    float acc[18];
#pragma unroll
    for (int d = 0; d < 18; ++d) acc[d] = b_off[d];
    for (int c = 0; c < Cc; ++c) {
        float xv = xb[(size_t)c * HWp + hw];
#pragma unroll
        for (int d = 0; d < 18; ++d) acc[d] += xv * w_off[d * Cc + c];
    }

    const float scale = 2.0f / 56.0f;
    float bx = (2.0f * w + 1.0f) / (float)Ww - 1.0f;
    float by = (2.0f * h + 1.0f) / (float)Hh - 1.0f;
    __bf16* dst = smp + (size_t)m * Kd;

    for (int f = 0; f < Ff; ++f) {
        // channel 2f = x-offset of filter f, 2f+1 = y-offset (reshape(B,F,2,H,W))
        float gx = bx + scale * acc[2 * f];
        float gy = by + scale * acc[2 * f + 1];
        float ix = ((gx + 1.0f) * Ww - 1.0f) * 0.5f;
        float iy = ((gy + 1.0f) * Hh - 1.0f) * 0.5f;
        float x0f = floorf(ix), y0f = floorf(iy);
        float wx1 = ix - x0f, wx0 = 1.0f - wx1;
        float wy1 = iy - y0f, wy0 = 1.0f - wy1;
        int x0 = (int)x0f, y0 = (int)y0f;
        int x1 = x0 + 1,   y1 = y0 + 1;
        float vx0 = (x0 >= 0 && x0 < Ww) ? 1.0f : 0.0f;
        float vx1 = (x1 >= 0 && x1 < Ww) ? 1.0f : 0.0f;
        float vy0 = (y0 >= 0 && y0 < Hh) ? 1.0f : 0.0f;
        float vy1 = (y1 >= 0 && y1 < Hh) ? 1.0f : 0.0f;
        int x0c = min(max(x0, 0), Ww - 1), x1c = min(max(x1, 0), Ww - 1);
        int y0c = min(max(y0, 0), Hh - 1), y1c = min(max(y1, 0), Hh - 1);
        float w00 = wy0 * wx0 * vy0 * vx0;
        float w01 = wy0 * wx1 * vy0 * vx1;
        float w10 = wy1 * wx0 * vy1 * vx0;
        float w11 = wy1 * wx1 * vy1 * vx1;
        int o00 = y0c * Ww + x0c, o01 = y0c * Ww + x1c;
        int o10 = y1c * Ww + x0c, o11 = y1c * Ww + x1c;

        const float* xc = xb;
#pragma unroll 4
        for (int c = 0; c < Cc; ++c) {
            float v = w00 * xc[o00] + w01 * xc[o01] + w10 * xc[o10] + w11 * xc[o11];
            dst[f * Cc + c] = (__bf16)v;
            xc += HWp;
        }
    }
}

// ---------------------------------------------------------------------------
// Kernel 3: bf16 WMMA GEMM  out[M, N] = sampled[M, K] * w_conv[N, K]^T + bias
// 128x128 tile / workgroup (8 waves), K-step 32, double-buffered LDS filled by
// GLOBAL_LOAD_ASYNC_TO_LDS_B128; per wave 2(M) x 4(N) v_wmma_f32_16x16x32_bf16.
// ---------------------------------------------------------------------------
#define MT  128
#define NT  128
#define KT  32
#define LDT 40                       // bf16 per LDS row: 32 + 8 pad (80B, 16B-aligned)
#define NKS (Kd / KT)                // 36
#define BUFSZ (2 * MT * LDT)         // one buffer: A tile + B tile

__global__ void __launch_bounds__(256)
gemm_wmma_kernel(const __bf16* __restrict__ A,   // [Mtot, Kd]
                 const __bf16* __restrict__ Wb,  // [OUTc, Kd]
                 const float* __restrict__ bias, // [OUTc]
                 float* __restrict__ out) {      // [B, OUT, H, W]
    __shared__ __align__(16) __bf16 lds[2 * BUFSZ];

    const int tid  = threadIdx.x;
    const int lane = tid & 31;
    const int wid  = tid >> 5;
    const int wm   = wid & 3;        // 32-row chunk within tile
    const int wn   = wid >> 2;       // 64-col chunk within tile
    const int m0   = blockIdx.x * MT;
    const int n0   = blockIdx.y * NT;

    v8f acc[2][4];
#pragma unroll
    for (int i = 0; i < 2; ++i)
#pragma unroll
        for (int j = 0; j < 4; ++j) acc[i][j] = (v8f)0.0f;

    // Tile pair = 1024 16B chunks (A: 0..511, B: 512..1023); 4 chunks / thread.
    auto copy_tile = [&](int buf, int ks) {
        const int k0 = ks * KT;
        __bf16* la = lds + buf * BUFSZ;
        __bf16* lb = la + MT * LDT;
#pragma unroll
        for (int q = 0; q < 4; ++q) {
            const int chunk = tid + q * 256;
            const int isB = chunk >> 9;
            const int c   = chunk & 511;
            const int r   = c >> 2;
            const int ci  = c & 3;
            const __bf16* g = (isB ? (Wb + (size_t)(n0 + r) * Kd)
                                   : (A  + (size_t)(m0 + r) * Kd)) + k0 + ci * 8;
            __bf16* l = (isB ? lb : la) + r * LDT + ci * 8;
            async_copy_b128(g, l);
        }
    };

    copy_tile(0, 0);
    wait_async_all();
    __syncthreads();

    const int kg = (lane >> 4) * 8;   // lanes 16-31 take K+8 halves (ISA A layout)
    const int rr = lane & 15;

    for (int ks = 0; ks < NKS; ++ks) {
        const int cur = ks & 1;
        if (ks + 1 < NKS) copy_tile(1 - cur, ks + 1);   // async fill of other buffer

        const __bf16* la = lds + cur * BUFSZ;
        const __bf16* lb = la + MT * LDT;

        v16bf afrag[2], bfrag[4];
#pragma unroll
        for (int i = 0; i < 2; ++i) {
            const __bf16* p = la + (wm * 32 + i * 16 + rr) * LDT;
            v8bf lo = *(const v8bf*)(p + kg);        // K = kg .. kg+7
            v8bf hi = *(const v8bf*)(p + kg + 16);   // K = kg+16 .. kg+23
            afrag[i] = __builtin_shufflevector(lo, hi, 0, 1, 2, 3, 4, 5, 6, 7,
                                               8, 9, 10, 11, 12, 13, 14, 15);
        }
#pragma unroll
        for (int j = 0; j < 4; ++j) {
            const __bf16* p = lb + (wn * 64 + j * 16 + rr) * LDT;
            v8bf lo = *(const v8bf*)(p + kg);
            v8bf hi = *(const v8bf*)(p + kg + 16);
            bfrag[j] = __builtin_shufflevector(lo, hi, 0, 1, 2, 3, 4, 5, 6, 7,
                                               8, 9, 10, 11, 12, 13, 14, 15);
        }
#pragma unroll
        for (int i = 0; i < 2; ++i)
#pragma unroll
            for (int j = 0; j < 4; ++j)
                acc[i][j] = __builtin_amdgcn_wmma_f32_16x16x32_bf16(
                    false, afrag[i], false, bfrag[j], (short)0, acc[i][j],
                    false, false);

        // Each wave drains its own ASYNCcnt, then the barrier publishes all
        // async LDS writes to the workgroup (CDNA5 async idiom).
        wait_async_all();
        __syncthreads();
    }

    // Epilogue: D layout -> lanes 0-15: N=lane, rows M=vgpr; lanes 16-31: M=vgpr+8
#pragma unroll
    for (int i = 0; i < 2; ++i) {
#pragma unroll
        for (int j = 0; j < 4; ++j) {
            const int nl = n0 + wn * 64 + j * 16 + (lane & 15);
            const float bv = bias[nl];
#pragma unroll
            for (int e = 0; e < 8; ++e) {
                const int ml = m0 + wm * 32 + i * 16 + e + ((lane >> 4) * 8);
                const int b  = ml / HWp;
                const int hw = ml - b * HWp;
                out[((size_t)b * OUTc + nl) * HWp + hw] = acc[i][j][e] + bv;
            }
        }
    }
}

// ---------------------------------------------------------------------------
extern "C" void kernel_launch(void* const* d_in, const int* in_sizes, int n_in,
                              void* d_out, int out_size, void* d_ws, size_t ws_size,
                              hipStream_t stream) {
    const float* x      = (const float*)d_in[0];  // [B,C,H,W]
    const float* w_off  = (const float*)d_in[1];  // [18, C]
    const float* b_off  = (const float*)d_in[2];  // [18]
    const float* w_conv = (const float*)d_in[3];  // [OUT, F*C]
    const float* b_conv = (const float*)d_in[4];  // [OUT]
    float* out = (float*)d_out;

    __bf16* wbf = (__bf16*)d_ws;                                   // 1.18 MB
    __bf16* smp = (__bf16*)((char*)d_ws + (size_t)OUTc * Kd * 2);  // 115.6 MB

    wconv_to_bf16_kernel<<<(OUTc * Kd + 255) / 256, 256, 0, stream>>>(
        w_conv, wbf, OUTc * Kd);

    deform_sample_kernel<<<Mtot / 256, 256, 0, stream>>>(x, w_off, b_off, smp);

    dim3 grid(Mtot / MT, OUTc / NT);   // 392 x 4
    gemm_wmma_kernel<<<grid, 256, 0, stream>>>(smp, wbf, b_conv, out);
}
